// PSCPAllChisNetwork_36215164240934
// MI455X (gfx1250) — matrix operand order repaired and checked by reading, not compiled, not run on hardware
//
#include <hip/hip_runtime.h>

// PSCP AllChis GVP-GNN forward for gfx1250 (MI455X).
// All matmuls (Wh/Ws/Wu of every GVP) run through v_wmma_f32_16x16x32_bf16.
// Vectors are stored plane-major [3*rows, ch] so the 'nci,hc->nhi' einsums
// become plain NT GEMMs. dst = repeat(arange(N),K) -> contiguous scatter-mean.
// GEMM: one wave owns a 32x(16*NV) super-tile (2xNV WMMA tiles), NV a template
// parameter so the inner loop is branch-free: one load clause, one wait, and
// 2*NV back-to-back v_wmma accumulates.

#define TPB 256
static const long NN = 16384;   // nodes
static const long KK = 30;      // top-k
static const long EE = NN * KK; // edges
#define GEPS 1e-8f

typedef unsigned short u16;
typedef __attribute__((ext_vector_type(16))) __bf16 v16bf;
typedef __attribute__((ext_vector_type(8)))  float  v8f;

static inline int pad16(int x){ return (x + 15) & ~15; }
static inline int pad32(int x){ return (x + 31) & ~31; }
static inline unsigned cdiv(long a, long b){ return (unsigned)((a + b - 1) / b); }

__device__ __forceinline__ u16 f2bf(float x){
  unsigned u = __float_as_uint(x);
  u += 0x7fffu + ((u >> 16) & 1u);              // round-nearest-even
  return (u16)(u >> 16);
}
__device__ __forceinline__ float bf2f(u16 h){
  return __uint_as_float(((unsigned)h) << 16);
}

// ---------------- WMMA GEMM:  C[M,Np] = A[M,Kp] * W[Np,Kp]^T + bias ----------------
// A,W bf16 row-major, lda/ldw multiples of 32; M mult of 32, Kp mult of 32.
// One wave computes a 32x(16*NV) super-tile; launch covers n-tiles
// [nt0base, nt0base + nsup*NV). No divergent or uniform branches inside.
template <int NV>
__global__ __launch_bounds__(TPB) void k_gemm_t(
    const u16* __restrict__ A, int lda,
    const u16* __restrict__ W, int ldw,
    const float* __restrict__ bias,
    long M, int nt0base, int nsup, int Kp,
    float* __restrict__ Cf, int ldcf,
    u16* __restrict__ Cb, int ldcb, int relu)
{
  int wv = threadIdx.x >> 5, lane = threadIdx.x & 31;
  long msup = M >> 5;                       // M multiple of 32
  long super = (long)blockIdx.x * 8 + wv;
  if (super >= msup * (long)nsup) return;
  long ms = super % msup;
  int  nt0 = nt0base + (int)(super / msup) * NV;
  int half = lane >> 4, l16 = lane & 15;
  // A fragment: lane holds row m = l16; elems e: K = 16*(e/8) + 8*half + e%8
  const u16* arow0 = A + (ms * 32 + l16) * (long)lda + 8 * half;
  const u16* arow1 = arow0 + 16 * (long)lda;
  // B fragment: lane holds col n = l16 (= weight row); K = 16*(e/8) + 8*half + e%8
  const u16* wrow[NV];
  #pragma unroll
  for (int j = 0; j < NV; ++j)
    wrow[j] = W + (long)((nt0 + j) * 16 + l16) * ldw + 8 * half;
  v8f acc[2][NV];
  #pragma unroll
  for (int mi = 0; mi < 2; ++mi)
    #pragma unroll
    for (int j = 0; j < NV; ++j) acc[mi][j] = {};
  for (int k0 = 0; k0 < Kp; k0 += 32) {
    union F { uint4 q[2]; v16bf v; };
    F a0, a1, b[NV];
    a0.q[0] = *(const uint4*)(arow0 + k0);
    a0.q[1] = *(const uint4*)(arow0 + k0 + 16);
    a1.q[0] = *(const uint4*)(arow1 + k0);
    a1.q[1] = *(const uint4*)(arow1 + k0 + 16);
    #pragma unroll
    for (int j = 0; j < NV; ++j) {
      b[j].q[0] = *(const uint4*)(wrow[j] + k0);
      b[j].q[1] = *(const uint4*)(wrow[j] + k0 + 16);
    }
    if (k0 + 32 < Kp) {
      __builtin_prefetch(arow0 + k0 + 32, 0, 1);
      __builtin_prefetch(arow1 + k0 + 32, 0, 1);
    }
    #pragma unroll
    for (int j = 0; j < NV; ++j) {
      acc[0][j] = __builtin_amdgcn_wmma_f32_16x16x32_bf16(false, a0.v, false, b[j].v,
                                                          (short)0, acc[0][j], false, false);
      acc[1][j] = __builtin_amdgcn_wmma_f32_16x16x32_bf16(false, a1.v, false, b[j].v,
                                                          (short)0, acc[1][j], false, false);
    }
  }
  #pragma unroll
  for (int j = 0; j < NV; ++j) {
    int ng = (nt0 + j) * 16 + l16;
    float bv = bias ? bias[ng] : 0.f;
    #pragma unroll
    for (int mi = 0; mi < 2; ++mi) {
      union { v8f v; float f[8]; } r; r.v = acc[mi][j];
      #pragma unroll
      for (int t = 0; t < 8; ++t) {          // C/D: m = 8*half + t, n = l16
        long mg = ms * 32 + mi * 16 + half * 8 + t;
        float v = r.f[t] + bv;
        if (relu) v = fmaxf(v, 0.f);
        if (Cf) Cf[mg * (long)ldcf + ng] = v;
        if (Cb) Cb[mg * (long)ldcb + ng] = f2bf(v);
      }
    }
  }
}

// -------- pack fp32 [R,C] (row stride lds) -> bf16 [Rp,ldp], zero padded --------
__global__ void k_pack_pad(u16* __restrict__ dst, const float* __restrict__ src,
                           long R, int C, int lds, long Rp, int ldp)
{
  long idx = (long)blockIdx.x * TPB + threadIdx.x;
  if (idx >= Rp * (long)ldp) return;
  long r = idx / ldp; int c = (int)(idx % ldp);
  float v = (r < R && c < C) ? src[r * (long)lds + c] : 0.f;
  dst[idx] = f2bf(v);
}

__global__ void k_pad_bias(float* dst, const float* src, int S, int Sp)
{
  int i = blockIdx.x * TPB + threadIdx.x;
  if (i >= Sp) return;
  dst[i] = (i < S) ? src[i] : 0.f;
}

// -------- fp32 [n, C, 3] -> bf16 plane-major [3*n, ld], zero padded --------
__global__ void k_pack_vec3(u16* __restrict__ dst, const float* __restrict__ src,
                            long n, int C, int ld)
{
  long idx = (long)blockIdx.x * TPB + threadIdx.x;
  if (idx >= 3 * n * (long)ld) return;
  long pr = idx / ld; int c = (int)(idx % ld);
  long i = pr / n, r = pr % n;
  float v = (c < C) ? src[(r * C + c) * 3 + i] : 0.f;
  dst[idx] = f2bf(v);
}

// -------- vn = sqrt(sum_i Vh_i^2 + eps) into cat cols [si, si+h) --------
__global__ void k_vn_fill(u16* __restrict__ cat, int ldcat, int si,
                          const u16* __restrict__ Vh, int ldvh, long n, int h)
{
  long idx = (long)blockIdx.x * TPB + threadIdx.x;
  if (idx >= n * (long)h) return;
  long r = idx / h; int c = (int)(idx % h);
  float s = GEPS;
  #pragma unroll
  for (int i = 0; i < 3; ++i) {
    float v = bf2f(Vh[((long)i * n + r) * ldvh + c]);
    s += v * v;
  }
  cat[r * (long)ldcat + si + c] = f2bf(sqrtf(s));
}

// -------- V *= sigmoid(||V||), fp32 in/out + optional bf16 out --------
__global__ void k_vgate(float* __restrict__ Vf, u16* __restrict__ Vb,
                        long n, int vo, int ldf, int ldb)
{
  long idx = (long)blockIdx.x * TPB + threadIdx.x;
  if (idx >= n * (long)vo) return;
  long r = idx / vo; int c = (int)(idx % vo);
  float v0 = Vf[(0 * n + r) * (long)ldf + c];
  float v1 = Vf[(1 * n + r) * (long)ldf + c];
  float v2 = Vf[(2 * n + r) * (long)ldf + c];
  float nrm = sqrtf(v0 * v0 + v1 * v1 + v2 * v2 + GEPS);
  float g = 1.f / (1.f + __expf(-nrm));
  v0 *= g; v1 *= g; v2 *= g;
  Vf[(0 * n + r) * (long)ldf + c] = v0;
  Vf[(1 * n + r) * (long)ldf + c] = v1;
  Vf[(2 * n + r) * (long)ldf + c] = v2;
  if (Vb) {
    Vb[(0 * n + r) * (long)ldb + c] = f2bf(v0);
    Vb[(1 * n + r) * (long)ldb + c] = f2bf(v1);
    Vb[(2 * n + r) * (long)ldb + c] = f2bf(v2);
  }
}

__global__ void k_copy_cols(u16* __restrict__ dst, int ldd,
                            const u16* __restrict__ src, int lds, long n, int C)
{
  long idx = (long)blockIdx.x * TPB + threadIdx.x;
  if (idx >= n * (long)C) return;
  long r = idx / C; int c = (int)(idx % C);
  dst[r * (long)ldd + c] = src[r * (long)lds + c];
}

// -------- edge scalar concat: [s[dst] | s[src] | es | (aa[src])] -> bf16 --------
__global__ void k_gather_scalar(u16* __restrict__ cat, int ldcat,
                                const float* __restrict__ s, int lds, int sdim,
                                const float* __restrict__ es, int ldes, int esdim,
                                const float* __restrict__ aat, const int* __restrict__ aaids,
                                const int* __restrict__ srcidx, long E)
{
  long idx = (long)blockIdx.x * TPB + threadIdx.x;
  if (idx >= E * (long)ldcat) return;
  long e = idx / ldcat; int c = (int)(idx % ldcat);
  float v = 0.f;
  if (c < sdim) {
    long d = e / KK; v = s[d * (long)lds + c];
  } else if (c < 2 * sdim) {
    int si = srcidx[e]; v = s[(long)si * lds + (c - sdim)];
  } else if (c < 2 * sdim + esdim) {
    v = es[e * (long)ldes + (c - 2 * sdim)];
  } else if (aat && c < 2 * sdim + esdim + 20) {
    int si = srcidx[e]; v = aat[aaids[si] * 20 + (c - 2 * sdim - esdim)];
  }
  cat[idx] = f2bf(v);
}

// -------- edge vector concat (plane-major): [V[dst] | V[src] | eV] -> bf16 --------
__global__ void k_gather_vec(u16* __restrict__ Vm, int ldm,
                             const float* __restrict__ Vn, int ldv, int vdim, long nn,
                             const float* __restrict__ eV, int ldev, int evdim,
                             const int* __restrict__ srcidx, long E)
{
  long idx = (long)blockIdx.x * TPB + threadIdx.x;
  if (idx >= 3 * E * (long)ldm) return;
  long pr = idx / ldm; int c = (int)(idx % ldm);
  long i = pr / E, e = pr % E;
  float v = 0.f;
  if (c < vdim) {
    long d = e / KK; v = Vn[(i * nn + d) * (long)ldv + c];
  } else if (c < 2 * vdim) {
    int si = srcidx[e]; v = Vn[(i * nn + si) * (long)ldv + (c - vdim)];
  } else if (c < 2 * vdim + evdim) {
    v = eV[(i * E + e) * (long)ldev + (c - 2 * vdim)];
  }
  Vm[idx] = f2bf(v);
}

// -------- SVLayerNorm (optionally with residual), one wave32 per row --------
__global__ __launch_bounds__(TPB) void k_svln(
    float* s_out, int ldso, u16* sb_out, int ldsb,
    float* v_out, int ldvo, u16* vb_out, int ldvb,
    const float* __restrict__ s_in, int ldsi, const float* __restrict__ s_res, int ldsr,
    const float* __restrict__ v_in, int ldvi, const float* __restrict__ v_res, int ldvr,
    long n, int Sd, int Vd)
{
  int lane = threadIdx.x & 31;
  long r = (long)blockIdx.x * 8 + (threadIdx.x >> 5);
  if (r >= n) return;
  int nc = Sd >> 5;                 // Sd in {64,128}
  float sv[4]; float sum = 0.f, sq = 0.f;
  for (int c = 0; c < nc; ++c) {
    int col = lane + (c << 5);
    float v = s_in[r * (long)ldsi + col];
    if (s_res) v += s_res[r * (long)ldsr + col];
    sv[c] = v; sum += v; sq += v * v;
  }
  for (int o = 16; o; o >>= 1) { sum += __shfl_xor(sum, o, 32); sq += __shfl_xor(sq, o, 32); }
  float mean = sum / Sd, var = sq / Sd - mean * mean;
  float inv = rsqrtf(var + 1e-5f);
  for (int c = 0; c < nc; ++c) {
    int col = lane + (c << 5);
    float v = (sv[c] - mean) * inv;
    if (s_out) s_out[r * (long)ldso + col] = v;
    if (sb_out) sb_out[r * (long)ldsb + col] = f2bf(v);
  }
  float vv[3] = {0.f, 0.f, 0.f}; float vsq = 0.f;
  if (lane < Vd) {
    #pragma unroll
    for (int i = 0; i < 3; ++i) {
      float v = v_in[((long)i * n + r) * ldvi + lane];
      if (v_res) v += v_res[((long)i * n + r) * ldvr + lane];
      vv[i] = v; vsq += v * v;
    }
  }
  for (int o = 16; o; o >>= 1) vsq += __shfl_xor(vsq, o, 32);
  float sc = rsqrtf(vsq / Vd + GEPS);
  #pragma unroll
  for (int i = 0; i < 3; ++i) {
    long po = (long)i * n + r;
    if (lane < Vd) {
      float v = vv[i] * sc;
      if (v_out) v_out[po * ldvo + lane] = v;
      if (vb_out) vb_out[po * ldvb + lane] = f2bf(v);
    } else if (vb_out && lane < ldvb) {
      vb_out[po * ldvb + lane] = 0;   // zero K-pad for next WMMA A operand
    }
  }
}

// -------- contiguous scatter-mean (K=30) + residual + SVLN, wave32 per node --------
__global__ __launch_bounds__(TPB) void k_update_svln(
    float* s_out, u16* sb_out, float* v_out, u16* vb_out,
    const float* __restrict__ s_old, const float* __restrict__ v_old,
    const float* __restrict__ ms, const float* __restrict__ mV, long n, long Etot)
{
  int lane = threadIdx.x & 31;
  long r = (long)blockIdx.x * 8 + (threadIdx.x >> 5);
  if (r >= n) return;
  long eb = r * KK;
  float acc[4] = {0.f, 0.f, 0.f, 0.f};
  for (int j = 0; j < (int)KK; ++j) {
    const float* row = ms + (eb + j) * 128;
    #pragma unroll
    for (int c = 0; c < 4; ++c) acc[c] += row[lane + (c << 5)];
  }
  float sv[4], sum = 0.f, sq = 0.f;
  #pragma unroll
  for (int c = 0; c < 4; ++c) {
    float v = s_old[r * 128 + lane + (c << 5)] + acc[c] * (1.f / KK);
    sv[c] = v; sum += v; sq += v * v;
  }
  for (int o = 16; o; o >>= 1) { sum += __shfl_xor(sum, o, 32); sq += __shfl_xor(sq, o, 32); }
  float mean = sum / 128.f, var = sq / 128.f - mean * mean;
  float inv = rsqrtf(var + 1e-5f);
  #pragma unroll
  for (int c = 0; c < 4; ++c) {
    int col = lane + (c << 5);
    float v = (sv[c] - mean) * inv;
    s_out[r * 128 + col] = v;
    sb_out[r * 128 + col] = f2bf(v);
  }
  float vv[3]; float vsq = 0.f;
  #pragma unroll
  for (int i = 0; i < 3; ++i) {
    const float* col = mV + ((long)i * Etot + eb) * 32 + lane;
    float a = 0.f;
    for (int j = 0; j < (int)KK; ++j) a += col[(long)j * 32];
    float v = v_old[((long)i * n + r) * 32 + lane] + a * (1.f / KK);
    vv[i] = v; vsq += v * v;
  }
  for (int o = 16; o; o >>= 1) vsq += __shfl_xor(vsq, o, 32);
  float sc = rsqrtf(vsq / 32.f + GEPS);
  #pragma unroll
  for (int i = 0; i < 3; ++i) {
    long po = (long)i * n + r;
    float v = vv[i] * sc;
    v_out[po * 32 + lane] = v;
    vb_out[po * 32 + lane] = f2bf(v);
  }
}

__global__ void k_chi_copy(float* __restrict__ dout, const float* __restrict__ chi,
                           int head, long n)
{
  long idx = (long)blockIdx.x * TPB + threadIdx.x;
  if (idx >= n * 72) return;
  long r = idx / 72; int c = (int)(idx % 72);
  dout[r * 288 + head * 72 + c] = chi[r * 80 + c];
}

__global__ void k_off_copy(float* __restrict__ dst, const float* __restrict__ src,
                           int head, long n)
{
  long i = (long)blockIdx.x * TPB + threadIdx.x;
  if (i >= n) return;
  dst[i * 4 + head] = src[i * 16];
}

// ================================= host =================================
struct GVPW {
  const float *Wh, *Ws, *Wu, *bs;
  int si, vi, so, vo, h;
  u16 *Whb, *Wsb, *Wub; float* bsp;
};

extern "C" void kernel_launch(void* const* d_in, const int* in_sizes, int n_in,
                              void* d_out, int out_size, void* d_ws, size_t ws_size,
                              hipStream_t stream)
{
  (void)in_sizes; (void)n_in; (void)out_size; (void)ws_size;
  const float* in_node_s = (const float*)d_in[0];   // [N,131]
  const float* in_node_v = (const float*)d_in[1];   // [N,3,3]
  const float* in_edge_s = (const float*)d_in[2];   // [E,64]
  const float* in_edge_v = (const float*)d_in[3];   // [E,4,3]
  const int*   edge_index = (const int*)d_in[4];    // [2,E]
  const int*   aa_ids = (const int*)d_in[5];        // [N]
  const int*   srcidx = edge_index + EE;

  // ---- params (jax sorted-key flatten: We0,We1,Wn0,Wn1,aa,dec,dense,enc,off,out;
  //      each GVP dict flattens as Wh,Ws,Wu,bs; layer dicts as f0,f1,m0,m1) ----
  int pi = 6;
  auto take = [&](int si, int vi, int so, int vo) {
    GVPW g{}; g.si = si; g.vi = vi; g.so = so; g.vo = vo; g.h = (vi > vo ? vi : vo);
    g.Wh = (const float*)d_in[pi]; g.Ws = (const float*)d_in[pi + 1];
    g.Wu = (const float*)d_in[pi + 2]; g.bs = (const float*)d_in[pi + 3];
    pi += 4; return g;
  };
  GVPW We0 = take(64, 4, 64, 16), We1 = take(64, 16, 64, 16);
  GVPW Wn0 = take(131, 3, 128, 32), Wn1 = take(128, 32, 128, 32);
  const float* aa_tab = (const float*)d_in[pi++];   // [20,20]
  struct LayerW { GVPW f0, f1, m0, m1; } decL[3], encL[3];
  for (int l = 0; l < 3; ++l) {
    decL[l].f0 = take(128, 32, 128, 32); decL[l].f1 = take(128, 32, 128, 32);
    decL[l].m0 = take(340, 80, 128, 32); decL[l].m1 = take(128, 32, 128, 32);
  }
  struct DenseW { GVPW d0, d1; } denseL[4];
  for (int d = 0; d < 4; ++d) { denseL[d].d0 = take(128, 32, 128, 32); denseL[d].d1 = take(128, 32, 72, 1); }
  for (int l = 0; l < 3; ++l) {
    encL[l].f0 = take(128, 32, 128, 32); encL[l].f1 = take(128, 32, 128, 32);
    encL[l].m0 = take(320, 80, 128, 32); encL[l].m1 = take(128, 32, 128, 32);
  }
  GVPW offw[4];
  for (int o = 0; o < 4; ++o) offw[o] = take(128, 32, 1, 1);
  GVPW Wout = take(128, 32, 128, 32);

  // ---- workspace bump allocator ----
  size_t wsoff = 0; char* wsb = (char*)d_ws;
  auto alloc = [&](size_t b) -> void* {
    void* p = wsb + wsoff; wsoff += (b + 255) & ~(size_t)255; return p;
  };

  auto pack = [&](GVPW& g) {
    int HP = pad16(g.h), KH = pad32(g.vi);
    int SOP = pad16(g.so), KS = pad32(g.si + g.h);
    int VOP = pad16(g.vo), KU = pad32(g.h);
    g.Whb = (u16*)alloc((size_t)HP * KH * 2);
    g.Wsb = (u16*)alloc((size_t)SOP * KS * 2);
    g.Wub = (u16*)alloc((size_t)VOP * KU * 2);
    g.bsp = (float*)alloc((size_t)SOP * 4);
    k_pack_pad<<<cdiv((long)HP * KH, TPB), TPB, 0, stream>>>(g.Whb, g.Wh, g.h, g.vi, g.vi, HP, KH);
    k_pack_pad<<<cdiv((long)SOP * KS, TPB), TPB, 0, stream>>>(g.Wsb, g.Ws, g.so, g.si + g.h, g.si + g.h, SOP, KS);
    k_pack_pad<<<cdiv((long)VOP * KU, TPB), TPB, 0, stream>>>(g.Wub, g.Wu, g.vo, g.h, g.h, VOP, KU);
    k_pad_bias<<<cdiv(SOP, TPB), TPB, 0, stream>>>(g.bsp, g.bs, g.so, SOP);
  };
  pack(We0); pack(We1); pack(Wn0); pack(Wn1);
  for (int l = 0; l < 3; ++l) { pack(decL[l].f0); pack(decL[l].f1); pack(decL[l].m0); pack(decL[l].m1); }
  for (int d = 0; d < 4; ++d) { pack(denseL[d].d0); pack(denseL[d].d1); }
  for (int l = 0; l < 3; ++l) { pack(encL[l].f0); pack(encL[l].f1); pack(encL[l].m0); pack(encL[l].m1); }
  for (int o = 0; o < 4; ++o) pack(offw[o]);
  pack(Wout);

  // ---- activation buffers ----
  float* node_s_f = (float*)alloc(NN * 128 * 4);
  float* nodeV_f  = (float*)alloc(3 * NN * 32 * 4);
  u16*   nodeV_b  = (u16*)alloc(3 * NN * 32 * 2);
  float* es_f     = (float*)alloc(EE * 64 * 4);
  float* eV_f     = (float*)alloc(3 * EE * 16 * 4);
  float* s2f = (float*)alloc(NN * 128 * 4);  u16* s2b = (u16*)alloc(NN * 128 * 2);
  float* V2f = (float*)alloc(3 * NN * 32 * 4); u16* V2b = (u16*)alloc(3 * NN * 32 * 2);
  u16* t_sb0 = (u16*)alloc(NN * 128 * 2);    u16* t_sb1 = (u16*)alloc(NN * 128 * 2);
  float* t_vf0 = (float*)alloc(3 * NN * 32 * 4); u16* t_vb0 = (u16*)alloc(3 * NN * 32 * 2);
  float* t_vf1 = (float*)alloc(3 * NN * 32 * 4); u16* t_vb1 = (u16*)alloc(3 * NN * 32 * 2);
  float* fsf = (float*)alloc(NN * 128 * 4);  float* fVf = (float*)alloc(3 * NN * 32 * 4);
  float* sFf = (float*)alloc(NN * 128 * 4);
  float* VFf = (float*)alloc(3 * NN * 32 * 4); u16* VFb = (u16*)alloc(3 * NN * 32 * 2);
  u16* catN = (u16*)alloc(NN * 192 * 2);
  u16* VhN  = (u16*)alloc(3 * NN * 32 * 2);
  float* chi_f = (float*)alloc(NN * 80 * 4);
  float* off_f = (float*)alloc(NN * 16 * 4);
  u16* catE  = (u16*)alloc(EE * 448 * 2);
  u16* VmE   = (u16*)alloc(3 * EE * 96 * 2);
  u16* VhE   = (u16*)alloc(3 * EE * 96 * 2);
  u16* msob  = (u16*)alloc(EE * 128 * 2);
  float* mvof = (float*)alloc(3 * EE * 32 * 4);
  u16* mvob  = (u16*)alloc(3 * EE * 32 * 2);
  u16* cat1E = (u16*)alloc(EE * 160 * 2);
  float* msE = (float*)alloc(EE * 128 * 4);
  float* mVE = (float*)alloc(3 * EE * 32 * 4);

  auto gemm = [&](const u16* A, int lda, const u16* W, int ldw, const float* bias,
                  long M, int Np, int Kp, float* Cf, int ldcf, u16* Cb, int ldcb, int relu) {
    long msup = M >> 5;
    int ntiles = Np >> 4, nfull = ntiles >> 2, rem = ntiles & 3;
    if (nfull)
      k_gemm_t<4><<<cdiv(msup * nfull, 8), TPB, 0, stream>>>(
          A, lda, W, ldw, bias, M, 0, nfull, Kp, Cf, ldcf, Cb, ldcb, relu);
    if (rem) {
      unsigned blocks = cdiv(msup, 8);
      int nt0 = nfull << 2;
      if (rem == 1)
        k_gemm_t<1><<<blocks, TPB, 0, stream>>>(A, lda, W, ldw, bias, M, nt0, 1, Kp, Cf, ldcf, Cb, ldcb, relu);
      else if (rem == 2)
        k_gemm_t<2><<<blocks, TPB, 0, stream>>>(A, lda, W, ldw, bias, M, nt0, 1, Kp, Cf, ldcf, Cb, ldcb, relu);
      else
        k_gemm_t<3><<<blocks, TPB, 0, stream>>>(A, lda, W, ldw, bias, M, nt0, 1, Kp, Cf, ldcf, Cb, ldcb, relu);
    }
  };

  // full GVP on n rows; cat cols [0, si) must be pre-filled by caller.
  auto run = [&](const GVPW& g, long n, const u16* Vin, u16* cat, u16* Vh,
                 float* so_f, u16* so_b, float* vo_f, u16* vo_b,
                 int sact, int vact, int skipv) {
    int HP16 = pad16(g.h), HP32 = pad32(g.h), KH = pad32(g.vi);
    int KS = pad32(g.si + g.h), SOP = pad16(g.so);
    if (HP32 != HP16) hipMemsetAsync(Vh, 0, (size_t)3 * n * HP32 * 2, stream);
    gemm(Vin, KH, g.Whb, KH, nullptr, 3 * n, HP16, KH, nullptr, 0, Vh, HP32, 0);
    k_vn_fill<<<cdiv(n * g.h, TPB), TPB, 0, stream>>>(cat, KS, g.si, Vh, HP32, n, g.h);
    gemm(cat, KS, g.Wsb, KS, g.bsp, n, SOP, KS, so_f, SOP, so_b, so_b ? pad32(g.so) : 0, sact);
    if (!skipv) {
      int VOP = pad16(g.vo);
      gemm(Vh, HP32, g.Wub, HP32, nullptr, 3 * n, VOP, HP32,
           vo_f, VOP, vact ? nullptr : vo_b, vo_b ? pad32(g.vo) : 0, 0);
      if (vact) k_vgate<<<cdiv(n * g.vo, TPB), TPB, 0, stream>>>(vo_f, vo_b, n, g.vo, VOP, pad32(g.vo));
    }
  };

  // ---------------- node stem: Wn0 -> svln -> Wn1 ----------------
  k_pack_vec3<<<cdiv(3 * NN * 32, TPB), TPB, 0, stream>>>(nodeV_b, in_node_v, NN, 3, 32);
  k_pack_pad<<<cdiv(NN * 192, TPB), TPB, 0, stream>>>(catN, in_node_s, NN, 131, 131, NN, 192);
  run(Wn0, NN, nodeV_b, catN, VhN, fsf, nullptr, fVf, nullptr, 0, 0, 0);
  k_svln<<<cdiv(NN, 8), TPB, 0, stream>>>(node_s_f, 128, nullptr, 0, nodeV_f, 32, nodeV_b, 32,
                                          fsf, 128, nullptr, 0, fVf, 32, nullptr, 0, NN, 128, 32);
  k_pack_pad<<<cdiv(NN * 160, TPB), TPB, 0, stream>>>(catN, node_s_f, NN, 128, 128, NN, 160);
  run(Wn1, NN, nodeV_b, catN, VhN, node_s_f, nullptr, nodeV_f, nullptr, 0, 0, 0);

  // ---------------- edge stem: We0 -> svln -> We1 ----------------
  k_pack_vec3<<<cdiv(3 * EE * 32, TPB), TPB, 0, stream>>>(mvob, in_edge_v, EE, 4, 32);
  k_pack_pad<<<cdiv(EE * 96, TPB), TPB, 0, stream>>>(catE, in_edge_s, EE, 64, 64, EE, 96);
  run(We0, EE, mvob, catE, VhE, msE, nullptr, mVE, nullptr, 0, 0, 0);
  k_svln<<<cdiv(EE, 8), TPB, 0, stream>>>(es_f, 64, nullptr, 0, eV_f, 16, mvob, 32,
                                          msE, 64, nullptr, 0, mVE, 16, nullptr, 0, EE, 64, 16);
  k_pack_pad<<<cdiv(EE * 96, TPB), TPB, 0, stream>>>(catE, es_f, EE, 64, 64, EE, 96);
  run(We1, EE, mvob, catE, VhE, es_f, nullptr, eV_f, nullptr, 0, 0, 0);

  // ---------------- conv layers ----------------
  auto conv = [&](LayerW& L, bool dec) {
    int si = dec ? 340 : 320;
    int ldc = pad32(si + 80);     // 448 dec / 416 enc
    k_gather_scalar<<<cdiv(EE * ldc, TPB), TPB, 0, stream>>>(
        catE, ldc, node_s_f, 128, 128, es_f, 64, 64,
        dec ? aa_tab : nullptr, aa_ids, srcidx, EE);
    k_gather_vec<<<cdiv(3 * EE * 96, TPB), TPB, 0, stream>>>(
        VmE, 96, nodeV_f, 32, 32, NN, eV_f, 16, 16, srcidx, EE);
    run(L.m0, EE, VmE, catE, VhE, nullptr, msob, mvof, mvob, 1, 1, 0);
    k_copy_cols<<<cdiv(EE * 128, TPB), TPB, 0, stream>>>(cat1E, 160, msob, 128, EE, 128);
    run(L.m1, EE, mvob, cat1E, VhE, msE, nullptr, mVE, nullptr, 0, 0, 0);
    k_update_svln<<<cdiv(NN, 8), TPB, 0, stream>>>(s2f, s2b, V2f, V2b,
                                                   node_s_f, nodeV_f, msE, mVE, NN, EE);
    k_pack_pad<<<cdiv(NN * 160, TPB), TPB, 0, stream>>>(catN, s2f, NN, 128, 128, NN, 160);
    run(L.f0, NN, V2b, catN, VhN, nullptr, t_sb1, t_vf0, t_vb0, 1, 1, 0);
    k_copy_cols<<<cdiv(NN * 128, TPB), TPB, 0, stream>>>(catN, 160, t_sb1, 128, NN, 128);
    run(L.f1, NN, t_vb0, catN, VhN, fsf, nullptr, fVf, nullptr, 0, 0, 0);
    k_svln<<<cdiv(NN, 8), TPB, 0, stream>>>(node_s_f, 128, nullptr, 0, nodeV_f, 32, nullptr, 0,
                                            fsf, 128, s2f, 128, fVf, 32, V2f, 32, NN, 128, 32);
  };
  for (int l = 0; l < 3; ++l) conv(encL[l], false);
  for (int l = 0; l < 3; ++l) conv(decL[l], true);

  // ---------------- heads ----------------
  k_svln<<<cdiv(NN, 8), TPB, 0, stream>>>(sFf, 128, nullptr, 0, VFf, 32, VFb, 32,
                                          node_s_f, 128, nullptr, 0, nodeV_f, 32, nullptr, 0, NN, 128, 32);
  k_pack_pad<<<cdiv(NN * 160, TPB), TPB, 0, stream>>>(catN, sFf, NN, 128, 128, NN, 160);
  run(Wout, NN, VFb, catN, VhN, nullptr, t_sb0, t_vf0, t_vb0, 1, 1, 0);

  float* dout = (float*)d_out;
  for (int d = 0; d < 4; ++d) {
    k_copy_cols<<<cdiv(NN * 128, TPB), TPB, 0, stream>>>(catN, 160, t_sb0, 128, NN, 128);
    run(denseL[d].d0, NN, t_vb0, catN, VhN, nullptr, t_sb1, t_vf1, t_vb1, 1, 1, 0);
    k_copy_cols<<<cdiv(NN * 128, TPB), TPB, 0, stream>>>(catN, 160, t_sb1, 128, NN, 128);
    run(denseL[d].d1, NN, t_vb1, catN, VhN, chi_f, nullptr, nullptr, nullptr, 0, 0, 1);
    k_chi_copy<<<cdiv(NN * 72, TPB), TPB, 0, stream>>>(dout, chi_f, d, NN);
  }
  float* dout_off = dout + NN * 288;
  for (int h = 0; h < 4; ++h) {
    k_copy_cols<<<cdiv(NN * 128, TPB), TPB, 0, stream>>>(catN, 160, t_sb0, 128, NN, 128);
    run(offw[h], NN, t_vb0, catN, VhN, off_f, nullptr, nullptr, nullptr, 0, 0, 1);
    k_off_copy<<<cdiv(NN, TPB), TPB, 0, stream>>>(dout_off, off_f, h, NN);
  }
}
